// AttentionGRUDecoder_12799002542536
// MI455X (gfx1250) — matrix operand order repaired and checked by reading, not compiled, not run on hardware
//
#include <hip/hip_runtime.h>
#include <hip/hip_bf16.h>

typedef __attribute__((ext_vector_type(16))) __bf16       v16bf;
typedef __attribute__((ext_vector_type(8)))  float        v8f;
typedef __attribute__((ext_vector_type(8)))  unsigned int v8u;

constexpr int B = 64, S = 196, E = 512, H = 512, A = 256, D = 512, V = 32000, L = 32;
constexpr int PAD_IDX = 2;

constexpr int NT    = 4;                 // N-tiles per wave (share one A fragment)
constexpr int WAVES = 8;                 // wave32 per 256-thread block
constexpr int NTILES_PER_BLK = NT * WAVES;   // 32 (512 columns per block)
constexpr int KMAX  = 1024;              // max K (gi GEMM); LDS A-tile = 32 KB

union FragBF { v8u u; v16bf b; };

__device__ __forceinline__ float sigmoidf_(float x) { return 1.0f / (1.0f + __expf(-x)); }

// ---------------------------------------------------------------------------
// WMMA GEMM:  C[m*ldc + n] = sum_k Amat[m*K+k] * Wmat[n*K+k] + bias[n]
//  - Amat (MxK bf16 row-major), Wmat (NxK bf16 row-major = B^T), C f32.
//  - Block owns one 16-row M tile + 32 N-tiles. The A tile (16 contiguous
//    rows = 32*K bytes) is staged in LDS ONCE per block with the CDNA5 async
//    path: global_load_async_to_lds_b128 + s_wait_asynccnt + barrier.
//  - Each wave computes NT=4 16x16 tiles per K-chunk from one shared A
//    fragment. All 4 B fragments are loaded into distinct registers BEFORE
//    the WMMA sequence so the loads issue as one clause and drain with
//    partial s_wait_loadcnt while the 4 WMMAs run back-to-back.
// Fragment layouts per CDNA5 ISA 7.12.2 (wave32):
//   A: lane m = lane%16, half = lane/16; two 16B runs at k = half*8, 16+half*8
//   B: lane n = lane%16, half = lane/16; two 16B runs at k = half*8, 16+half*8
//   C/D: VGPR r -> (M = half*8 + r, N = lane%16)
// ---------------------------------------------------------------------------
__global__ void __launch_bounds__(256)
wmma_gemm_bf16(const __bf16* __restrict__ amat,
               const __bf16* __restrict__ wmat,
               const float*  __restrict__ bias,
               float*        __restrict__ cmat,
               int M, int N, int K, int ldc)
{
    __shared__ __align__(16) __bf16 aTile[16 * KMAX];   // 32 KB LDS

    const int ntiles  = N >> 4;
    const int ngroups = (ntiles + NTILES_PER_BLK - 1) / NTILES_PER_BLK;
    const int mtile   = blockIdx.x / ngroups;
    const int ngroup  = blockIdx.x - mtile * ngroups;

    // ---- async fill of the A tile (rows are contiguous: one flat 32*K-byte copy)
    const unsigned ldsBase = (unsigned)(size_t)(&aTile[0]);   // low 32 bits = LDS offset
    const __bf16*  abase   = amat + (size_t)(mtile * 16) * K;
    const int nvec = (16 * K) >> 3;                           // # of 16-byte chunks
    for (int i = threadIdx.x; i < nvec; i += blockDim.x) {
        unsigned           lds = ldsBase + (unsigned)(i * 16);
        unsigned long long ga  = (unsigned long long)(size_t)abase
                               + (unsigned long long)i * 16u;
        asm volatile("global_load_async_to_lds_b128 %0, %1, off"
                     :: "v"(lds), "v"(ga) : "memory");
    }
    asm volatile("s_wait_asynccnt 0" ::: "memory");
    __syncthreads();

    const int wave = threadIdx.x >> 5;
    const int lane = threadIdx.x & 31;
    const int half = lane >> 4;
    const int mn   = lane & 15;

    const int nt0 = ngroup * NTILES_PER_BLK + wave * NT;
    if (nt0 >= ntiles) return;              // wave-uniform: live waves keep EXEC all-1s

    const __bf16* lrow = aTile + mn * K;    // this lane's A row in LDS

    const __bf16* wrow[NT];
    bool          valid[NT];
#pragma unroll
    for (int j = 0; j < NT; ++j) {
        int nt   = nt0 + j;
        valid[j] = (nt < ntiles);
        wrow[j]  = wmat + (size_t)((valid[j] ? nt : nt0) * 16 + mn) * K;
    }

    v8f acc[NT] = {};
    for (int kc = 0; kc < K; kc += 32) {
        // issue every load for this K-chunk up front (2x ds_load_b128 + 8x global_load_b128)
        uint4 a0 = *(const uint4*)(lrow + kc + half * 8);
        uint4 a1 = *(const uint4*)(lrow + kc + 16 + half * 8);
        uint4 pb0[NT], pb1[NT];
#pragma unroll
        for (int j = 0; j < NT; ++j) {
            pb0[j] = *(const uint4*)(wrow[j] + kc + half * 8);
            pb1[j] = *(const uint4*)(wrow[j] + kc + 16 + half * 8);
        }
        FragBF fa;
        fa.u[0] = a0.x; fa.u[1] = a0.y; fa.u[2] = a0.z; fa.u[3] = a0.w;
        fa.u[4] = a1.x; fa.u[5] = a1.y; fa.u[6] = a1.z; fa.u[7] = a1.w;
#pragma unroll
        for (int j = 0; j < NT; ++j) {
            FragBF fb;
            fb.u[0] = pb0[j].x; fb.u[1] = pb0[j].y; fb.u[2] = pb0[j].z; fb.u[3] = pb0[j].w;
            fb.u[4] = pb1[j].x; fb.u[5] = pb1[j].y; fb.u[6] = pb1[j].z; fb.u[7] = pb1[j].w;
            acc[j] = __builtin_amdgcn_wmma_f32_16x16x32_bf16(
                /*neg_a=*/false, fa.b, /*neg_b=*/false, fb.b,
                /*c_mod=*/(short)0, acc[j], /*reuse_a=*/false, /*reuse_b=*/false);
        }
    }

#pragma unroll
    for (int j = 0; j < NT; ++j) {
        if (!valid[j]) continue;
        const int   n  = (nt0 + j) * 16 + mn;
        const float bv = bias ? bias[n] : 0.0f;
        float* crow = cmat + (size_t)(mtile * 16 + half * 8) * ldc + n;
#pragma unroll
        for (int r = 0; r < 8; ++r) crow[(size_t)r * ldc] = acc[j][r] + bv;
    }
}

// ---------------------------------------------------------------------------
__global__ void cvt_f32_to_bf16(const float* __restrict__ src,
                                __bf16* __restrict__ dst, long n)
{
    long i = (long)blockIdx.x * blockDim.x + threadIdx.x;
    if (i < n) dst[i] = (__bf16)src[i];
}

// h0 = tanh(pooled @ W_init^T + b_init); one thread per (b, j)
__global__ void h0_kernel(const float* __restrict__ pooled,
                          const float* __restrict__ w_init,
                          const float* __restrict__ b_init,
                          float* __restrict__ h, __bf16* __restrict__ hb)
{
    int idx = blockIdx.x * blockDim.x + threadIdx.x;   // B*H threads
    int b = idx / H, j = idx - b * H;
    const float* p = pooled + (size_t)b * E;
    const float* w = w_init + (size_t)j * E;
    float acc = b_init[j];
    for (int e = 0; e < E; ++e) acc += p[e] * w[e];
    float v = tanhf(acc);
    h[idx]  = v;
    hb[idx] = (__bf16)v;
}

// Per-batch attention: scores -> softmax -> context, plus x = [embed, context] bf16
__global__ void attn_kernel(const float* __restrict__ encp,      // (B,S,A)
                            const float* __restrict__ decp,      // (B,A)
                            const float* __restrict__ wscore,    // (A)
                            const float* __restrict__ bscore,    // (1)
                            const float* __restrict__ enc,       // (B,S,E)
                            const float* __restrict__ emb,       // (V,D)
                            const int*   __restrict__ targets,   // (B,L)
                            int t,
                            __bf16* __restrict__ xb)             // (B, D+E)
{
    const int b = blockIdx.x;
    __shared__ float sc[S];
    __shared__ float inv_sum;

    const float* ep_base = encp + (size_t)b * S * A;
    const float* dp      = decp + (size_t)b * A;

    for (int s = threadIdx.x; s < S; s += blockDim.x) {
        const float* ep = ep_base + (size_t)s * A;
        float acc = 0.0f;
        for (int a = 0; a < A; ++a) acc += tanhf(ep[a] + dp[a]) * wscore[a];
        sc[s] = acc + bscore[0];
    }
    __syncthreads();

    if (threadIdx.x == 0) {
        float mx = -3.4e38f;
        for (int s = 0; s < S; ++s) mx = fmaxf(mx, sc[s]);
        float sum = 0.0f;
        for (int s = 0; s < S; ++s) { float e = __expf(sc[s] - mx); sc[s] = e; sum += e; }
        inv_sum = 1.0f / sum;
    }
    __syncthreads();
    const float inv = inv_sum;

    // embedded half of x (padding_idx -> zero row)
    const int tok = targets[b * L + t];
    for (int d = threadIdx.x; d < D; d += blockDim.x) {
        float v = (tok == PAD_IDX) ? 0.0f : emb[(size_t)tok * D + d];
        xb[(size_t)b * (D + E) + d] = (__bf16)v;
    }
    // context half of x
    const float* eb = enc + (size_t)b * S * E;
    for (int e = threadIdx.x; e < E; e += blockDim.x) {
        float acc = 0.0f;
        for (int s = 0; s < S; ++s) acc += sc[s] * eb[(size_t)s * E + e];
        xb[(size_t)b * (D + E) + D + e] = (__bf16)(acc * inv);
    }
}

// GRU gate math (PyTorch r,z,n ordering); updates h in place, mirrors to bf16
__global__ void gru_kernel(const float* __restrict__ gi,   // (B,3H)
                           const float* __restrict__ gh,   // (B,3H)
                           float* __restrict__ h,          // (B,H) in/out
                           __bf16* __restrict__ hb)        // (B,H) out
{
    int idx = blockIdx.x * blockDim.x + threadIdx.x;        // B*H threads
    int b = idx / H, j = idx - b * H;
    const float* gib = gi + (size_t)b * 3 * H;
    const float* ghb = gh + (size_t)b * 3 * H;
    float r  = sigmoidf_(gib[j]         + ghb[j]);
    float z  = sigmoidf_(gib[H + j]     + ghb[H + j]);
    float n  = tanhf    (gib[2 * H + j] + r * ghb[2 * H + j]);
    float hn = (1.0f - z) * n + z * h[idx];
    h[idx]  = hn;
    hb[idx] = (__bf16)hn;
}

// ---------------------------------------------------------------------------
extern "C" void kernel_launch(void* const* d_in, const int* in_sizes, int n_in,
                              void* d_out, int out_size, void* d_ws, size_t ws_size,
                              hipStream_t stream)
{
    (void)in_sizes; (void)n_in; (void)out_size; (void)ws_size;

    const float* encoder_out = (const float*)d_in[0];
    const float* pooled      = (const float*)d_in[1];
    const int*   targets     = (const int*)  d_in[2];
    const float* embedding   = (const float*)d_in[3];
    const float* W_enc       = (const float*)d_in[4];
    const float* b_enc       = (const float*)d_in[5];
    const float* W_dec       = (const float*)d_in[6];
    const float* b_dec       = (const float*)d_in[7];
    const float* W_score     = (const float*)d_in[8];
    const float* b_score     = (const float*)d_in[9];
    const float* W_ih        = (const float*)d_in[10];
    const float* W_hh        = (const float*)d_in[11];
    const float* b_ih        = (const float*)d_in[12];
    const float* b_hh        = (const float*)d_in[13];
    const float* W_out       = (const float*)d_in[14];
    const float* b_out       = (const float*)d_in[15];
    const float* W_init      = (const float*)d_in[16];
    const float* b_init      = (const float*)d_in[17];
    float* out = (float*)d_out;

    // --- workspace carve-out (~65 MB) ---
    char* ws = (char*)d_ws;
    size_t off = 0;
    auto carve = [&](size_t bytes) -> char* {
        char* p = ws + off; off += (bytes + 255) & ~(size_t)255; return p;
    };
    __bf16* enc_b  = (__bf16*)carve((size_t)B * S * E * 2);
    float*  encp   = (float*) carve((size_t)B * S * A * 4);
    __bf16* Wout_b = (__bf16*)carve((size_t)V * H * 2);
    __bf16* Wih_b  = (__bf16*)carve((size_t)3 * H * (D + E) * 2);
    __bf16* Whh_b  = (__bf16*)carve((size_t)3 * H * H * 2);
    __bf16* Wdec_b = (__bf16*)carve((size_t)A * H * 2);
    __bf16* Wenc_b = (__bf16*)carve((size_t)A * E * 2);
    float*  hbuf   = (float*) carve((size_t)B * H * 4);
    __bf16* h_b    = (__bf16*)carve((size_t)B * H * 2);
    float*  decp   = (float*) carve((size_t)B * A * 4);
    __bf16* x_b    = (__bf16*)carve((size_t)B * (D + E) * 2);
    float*  gi     = (float*) carve((size_t)B * 3 * H * 4);
    float*  gh     = (float*) carve((size_t)B * 3 * H * 4);

    auto cvt = [&](const float* s, __bf16* d, long n) {
        cvt_f32_to_bf16<<<dim3((unsigned)((n + 255) / 256)), dim3(256), 0, stream>>>(s, d, n);
    };
    auto gemm = [&](const __bf16* am, const __bf16* wm, const float* bias,
                    float* cm, int M, int N, int K, int ldc) {
        int mtiles  = M / 16;
        int ntiles  = N / 16;
        int ngroups = (ntiles + NTILES_PER_BLK - 1) / NTILES_PER_BLK;
        wmma_gemm_bf16<<<dim3(mtiles * ngroups), dim3(256), 0, stream>>>(
            am, wm, bias, cm, M, N, K, ldc);
    };

    // one-time bf16 conversions (weights stay L2-resident across the 31 steps)
    cvt(encoder_out, enc_b,  (long)B * S * E);
    cvt(W_out,       Wout_b, (long)V * H);
    cvt(W_ih,        Wih_b,  (long)3 * H * (D + E));
    cvt(W_hh,        Whh_b,  (long)3 * H * H);
    cvt(W_dec,       Wdec_b, (long)A * H);
    cvt(W_enc,       Wenc_b, (long)A * E);

    // h0 = tanh(pooled @ W_init^T + b_init)
    h0_kernel<<<dim3(B * H / 256), dim3(256), 0, stream>>>(pooled, W_init, b_init, hbuf, h_b);

    // enc_proj = encoder_out @ W_enc^T + b_enc   (hoisted, step-invariant)
    gemm(enc_b, Wenc_b, b_enc, encp, B * S, A, E, A);

    for (int t = 0; t < L - 1; ++t) {
        // dec_proj = h @ W_dec^T + b_dec
        gemm(h_b, Wdec_b, b_dec, decp, B, A, H, A);
        // attention + softmax + context + x concat
        attn_kernel<<<dim3(B), dim3(256), 0, stream>>>(
            encp, decp, W_score, b_score, encoder_out, embedding, targets, t, x_b);
        // gi = x @ W_ih^T + b_ih ; gh = h @ W_hh^T + b_hh
        gemm(x_b, Wih_b, b_ih, gi, B, 3 * H, D + E, 3 * H);
        gemm(h_b, Whh_b, b_hh, gh, B, 3 * H, H, 3 * H);
        // GRU gates -> h (in place) + bf16 mirror
        gru_kernel<<<dim3(B * H / 256), dim3(256), 0, stream>>>(gi, gh, hbuf, h_b);
        // logits = h_new @ W_out^T + b_out -> d_out[:, t, :]
        gemm(h_b, Wout_b, b_out, out + (size_t)t * V, B, V, H, (L - 1) * V);
    }
}
